// TemporalAlignment_48868137894265
// MI455X (gfx1250) — compile-verified, matrix operand order checked by reading, not run on hardware
//
#include <hip/hip_runtime.h>

// Problem constants (from reference)
#define BB      2
#define NN      2048
#define DIMM    2048
#define HEADS_  16
#define DHEAD   64
#define INNER_  1024
#define MTOT    (BB * NN)        // 4096 flattened rows
#define SCALE_F 0.125f           // 64^-0.5

typedef __attribute__((ext_vector_type(16))) _Float16 v16h;
typedef __attribute__((ext_vector_type(8)))  _Float16 v8h;
typedef __attribute__((ext_vector_type(4)))  _Float16 v4h;
typedef __attribute__((ext_vector_type(8)))  float    v8f;

// ---------------------------------------------------------------------------
// CDNA5 async data movement (CDNA5 ISA): ASYNCcnt-tracked DMA of
// 16 bytes from global to LDS, no VGPR data. LDS address = low 32 bits of the
// flat shared-aperture address.
// ---------------------------------------------------------------------------
__device__ __forceinline__ unsigned lds_off(const void* p) {
  return (unsigned)(unsigned long long)p;
}
__device__ __forceinline__ void async_copy_b128(unsigned lds_addr, const void* gptr) {
  asm volatile("global_load_async_to_lds_b128 %0, %1, off"
               :: "v"(lds_addr), "v"((unsigned long long)gptr)
               : "memory");
}
template <int N>
__device__ __forceinline__ void wait_async() {
  asm volatile("s_wait_asynccnt %0" :: "i"(N) : "memory");
}

// CDNA5 f16 A/B fragment: lane half kb = 0 (lanes 0-15) or 8 (lanes 16-31);
// lane holds halves k = kb..kb+7 and k = 16+kb..16+kb+7 -> two aligned 16B runs.
__device__ __forceinline__ v16h load_frag(const _Float16* row, int kb) {
  const v8h lo = *reinterpret_cast<const v8h*>(row + kb);
  const v8h hi = *reinterpret_cast<const v8h*>(row + 16 + kb);
  return __builtin_shufflevector(lo, hi, 0, 1, 2, 3, 4, 5, 6, 7,
                                 8, 9, 10, 11, 12, 13, 14, 15);
}

#define WMMA_F16(Af, Bf, Cf) \
  __builtin_amdgcn_wmma_f32_16x16x32_f16(false, (Af), false, (Bf), (short)0, (Cf), false, false)

// ---------------------------------------------------------------------------
// Prep kernels (bandwidth-bound, run once per call):
// fp32 -> f16 elementwise, and fp32 [K][N] -> f16 [N][K] transpose-convert.
// ---------------------------------------------------------------------------
__global__ __launch_bounds__(256)
void ta_cvt_f32_f16(const float* __restrict__ X, _Float16* __restrict__ Y, int n4)
{
  const int i = blockIdx.x * 256 + threadIdx.x;
  if (i < n4) {
    const float4 v = reinterpret_cast<const float4*>(X)[i];
    const v4h h = { (_Float16)v.x, (_Float16)v.y, (_Float16)v.z, (_Float16)v.w };
    reinterpret_cast<v4h*>(Y)[i] = h;
  }
}

__global__ __launch_bounds__(256)
void ta_transpose_cvt(const float* __restrict__ W, _Float16* __restrict__ Wt,
                      int K, int N)   // W: [K][N] fp32 -> Wt: [N][K] f16
{
  __shared__ _Float16 t[64][68];
  const int tid = threadIdx.x;
  const int k0 = blockIdx.y << 6, n0 = blockIdx.x << 6;
  for (int i = tid; i < 1024; i += 256) {           // 64 rows x 16 float4
    const int r = i >> 4, c4 = (i & 15) << 2;
    const float4 v = *reinterpret_cast<const float4*>(W + (size_t)(k0 + r) * N + n0 + c4);
    t[r][c4 + 0] = (_Float16)v.x; t[r][c4 + 1] = (_Float16)v.y;
    t[r][c4 + 2] = (_Float16)v.z; t[r][c4 + 3] = (_Float16)v.w;
  }
  __syncthreads();
  for (int i = tid; i < 1024; i += 256) {
    const int n = i >> 4, k4 = (i & 15) << 2;
    const v4h h = { t[k4 + 0][n], t[k4 + 1][n], t[k4 + 2][n], t[k4 + 3][n] };
    *reinterpret_cast<v4h*>(&Wt[(size_t)(n0 + n) * K + k0 + k4]) = h;
  }
}

// ---------------------------------------------------------------------------
// WMMA GEMM, all-f16 operands: C[M,N] = A[M,K] * Bt[N,K]^T  (f32 accumulate)
// Block tile 128x128, BK=32, 512 threads = 16 waves, each wave 2x2 16x16 tiles.
// Staging is pure global_load_async_to_lds_b128 (one per thread per operand
// per K-step), double-buffered and branch-free: the pipeline loop is manually
// unrolled by 2 so the buffer index is a compile-time constant.
// If Ch != nullptr -> f16 output; else fp32 with optional residual and bias.
// K must be a multiple of 64.
// ---------------------------------------------------------------------------
__global__ __launch_bounds__(512)
void ta_gemm_wmma_f16(const _Float16* __restrict__ A,   // [M][K]
                      const _Float16* __restrict__ Bt,  // [N][K]
                      _Float16* __restrict__ Ch, float* __restrict__ Cf,
                      const float* __restrict__ resid, const float* __restrict__ bias,
                      int M, int N, int K)
{
  __shared__ _Float16 As[2][128][48];   // [m][k], 96B row stride (16B multiple)
  __shared__ _Float16 Bs[2][128][48];   // [n][k]

  const int tid  = threadIdx.x;
  const int lane = tid & 31;
  const int wave = tid >> 5;              // 0..15
  const int wm   = wave >> 2;             // 0..3 (32-row strip)
  const int wn   = wave & 3;              // 0..3 (32-col strip)
  const int bm   = blockIdx.y << 7;
  const int bn   = blockIdx.x << 7;
  const int ml   = lane & 15;
  const int kb   = (lane & 16) ? 8 : 0;   // frag K base AND C/D row offset

  // Per-thread staging slot: one b128 (8 halves) of A and of B per K-step.
  const int r_st  = tid >> 2;             // 0..127
  const int c8_st = (tid & 3) << 3;       // 0,8,16,24
  const _Float16* gA = A  + (size_t)(bm + r_st) * K + c8_st;
  const _Float16* gB = Bt + (size_t)(bn + r_st) * K + c8_st;
  const unsigned ldsA[2] = { lds_off(&As[0][r_st][c8_st]), lds_off(&As[1][r_st][c8_st]) };
  const unsigned ldsB[2] = { lds_off(&Bs[0][r_st][c8_st]), lds_off(&Bs[1][r_st][c8_st]) };

  v8f acc[2][2] = {};

  auto issue = [&](int b2, int kkoff) {
    async_copy_b128(ldsA[b2], gA + kkoff);
    async_copy_b128(ldsB[b2], gB + kkoff);
  };
  auto compute = [&](int b2) {
    const v16h a0 = load_frag(&As[b2][wm * 32      + ml][0], kb);
    const v16h a1 = load_frag(&As[b2][wm * 32 + 16 + ml][0], kb);
    const v16h b0 = load_frag(&Bs[b2][wn * 32      + ml][0], kb);
    const v16h b1 = load_frag(&Bs[b2][wn * 32 + 16 + ml][0], kb);
    acc[0][0] = WMMA_F16(a0, b0, acc[0][0]);
    acc[0][1] = WMMA_F16(a0, b1, acc[0][1]);
    acc[1][0] = WMMA_F16(a1, b0, acc[1][0]);
    acc[1][1] = WMMA_F16(a1, b1, acc[1][1]);
  };

  // Software pipeline: tile t lives in buffer t&1; in-order ASYNCcnt completion
  // means "wait <= 2" after issuing the next tile guarantees the current tile
  // has fully landed in LDS.
  issue(0, 0);
  int kk = 0;
  for (; kk + 64 < K; kk += 64) {
    issue(1, kk + 32);
    wait_async<2>();
    __syncthreads();
    compute(0);
    __syncthreads();
    issue(0, kk + 64);
    wait_async<2>();
    __syncthreads();
    compute(1);
    __syncthreads();
  }
  issue(1, kk + 32);                       // last tile
  wait_async<2>();
  __syncthreads();
  compute(0);
  __syncthreads();
  wait_async<0>();
  __syncthreads();
  compute(1);

  // C/D layout: reg r -> M = r + 8*(lane>=16) (== kb), N = lane&15
  #pragma unroll
  for (int tr = 0; tr < 2; ++tr) {
    #pragma unroll
    for (int tc = 0; tc < 2; ++tc) {
      const int mrow0 = bm + wm * 32 + tr * 16 + kb;
      const int col   = bn + wn * 32 + tc * 16 + ml;
      if (Ch) {
        #pragma unroll
        for (int r = 0; r < 8; ++r)
          Ch[(size_t)(mrow0 + r) * N + col] = (_Float16)acc[tr][tc][r];
      } else {
        #pragma unroll
        for (int r = 0; r < 8; ++r) {
          const size_t idx = (size_t)(mrow0 + r) * N + col;
          float v = acc[tr][tc][r];
          if (resid) v += resid[idx];
          if (bias)  v += bias[col];
          Cf[idx] = v;
        }
      }
    }
  }
}

// ---------------------------------------------------------------------------
// Flash attention over one (batch, head): block = 16 query rows, 4 waves.
// Wave w owns key-cols / out-dims [w*16, w*16+16). K tile staged via
// async-to-LDS DMA; V tile transposed into LDS manually (overlaps the DMA).
// ---------------------------------------------------------------------------
__global__ __launch_bounds__(128)
void ta_flash_attn_wmma(const _Float16* __restrict__ Q, const _Float16* __restrict__ Km,
                        const _Float16* __restrict__ Vm, float* __restrict__ O,
                        int accum, float scale)
{
  __shared__ _Float16 Ks [64][72];   // [key][dim], 144B stride (16B multiple)
  __shared__ _Float16 VsT[64][72];   // [dim][key]
  __shared__ _Float16 Ps [16][72];   // [m][key]
  __shared__ float stepmax[4][16];
  __shared__ float stepsum[4][16];
  __shared__ float Mrow[16], Lrow[16], scf[16];

  const int tid  = threadIdx.x;
  const int lane = tid & 31;
  const int w    = tid >> 5;              // 0..3
  const int ml   = lane & 15;
  const int kb   = (lane & 16) ? 8 : 0;
  const int q0r  = blockIdx.x << 4;
  const int head = blockIdx.y;
  const int b    = blockIdx.z;
  const size_t base = ((size_t)b * NN) * INNER_ + (size_t)head * DHEAD;

  // Q fragments once: global_load_b128 pairs
  const _Float16* qrow = Q + base + (size_t)(q0r + ml) * INNER_;
  const v16h qa0 = load_frag(qrow, kb);
  const v16h qa1 = load_frag(qrow + 32, kb);

  if (tid < 16) { Mrow[tid] = -1e30f; Lrow[tid] = 0.0f; }
  v8f o = {};
  __syncthreads();

  for (int j0 = 0; j0 < NN; j0 += 64) {
    // Async-DMA the K tile (64 keys x 64 dims f16 = 512 b128s, 4 per thread)
    #pragma unroll
    for (int i = 0; i < 4; ++i) {
      const int idx = tid + i * 128;            // 0..511
      const int r = idx >> 3, c8 = (idx & 7) << 3;
      async_copy_b128(lds_off(&Ks[r][c8]),
                      Km + base + (size_t)(j0 + r) * INNER_ + c8);
    }
    // Transposed V staging (overlaps with the async K DMA in flight)
    for (int i = tid; i < 1024; i += 128) {
      const int r = i >> 4, c = (i & 15) << 2;  // key r, dim c..c+3
      const v4h vv = *reinterpret_cast<const v4h*>(Vm + base + (size_t)(j0 + r) * INNER_ + c);
      VsT[c + 0][r] = vv[0];
      VsT[c + 1][r] = vv[1];
      VsT[c + 2][r] = vv[2];
      VsT[c + 3][r] = vv[3];
    }
    wait_async<0>();
    __syncthreads();

    // S = scale * Q K^T : B[k=dim][n=key] = Ks[key][dim] (K-contiguous)
    const v16h kb0 = load_frag(&Ks[w * 16 + ml][0],  kb);
    const v16h kb1 = load_frag(&Ks[w * 16 + ml][32], kb);
    v8f s = {};
    s = WMMA_F16(qa0, kb0, s);
    s = WMMA_F16(qa1, kb1, s);

    // Per-row max of this wave's 16 key columns (reduce across the 16 N lanes)
    #pragma unroll
    for (int r = 0; r < 8; ++r) {
      s[r] *= scale;
      float v = s[r];
      v = fmaxf(v, __shfl_xor(v, 1, 32));
      v = fmaxf(v, __shfl_xor(v, 2, 32));
      v = fmaxf(v, __shfl_xor(v, 4, 32));
      v = fmaxf(v, __shfl_xor(v, 8, 32));
      if (ml == 0) stepmax[w][kb + r] = v;
    }
    __syncthreads();
    if (tid < 16) {
      const float sm = fmaxf(fmaxf(stepmax[0][tid], stepmax[1][tid]),
                             fmaxf(stepmax[2][tid], stepmax[3][tid]));
      const float mn = fmaxf(Mrow[tid], sm);
      scf[tid]  = __expf(Mrow[tid] - mn);
      Mrow[tid] = mn;
    }
    __syncthreads();

    // P = exp(S - Mrow): stash to LDS f16, per-row partial sums, rescale O
    #pragma unroll
    for (int r = 0; r < 8; ++r) {
      const int m = kb + r;
      const float p = __expf(s[r] - Mrow[m]);
      Ps[m][w * 16 + ml] = (_Float16)p;
      float v = p;
      v += __shfl_xor(v, 1, 32);
      v += __shfl_xor(v, 2, 32);
      v += __shfl_xor(v, 4, 32);
      v += __shfl_xor(v, 8, 32);
      if (ml == 0) stepsum[w][m] = v;
      o[r] *= scf[m];
    }
    __syncthreads();
    if (tid < 16)
      Lrow[tid] = Lrow[tid] * scf[tid]
                + stepsum[0][tid] + stepsum[1][tid] + stepsum[2][tid] + stepsum[3][tid];

    // O += P V : A = Ps[m][key], B[k=key][n=dim] = VsT[dim][key] (both K-contig)
    const v16h pa0 = load_frag(&Ps[ml][0],  kb);
    const v16h pa1 = load_frag(&Ps[ml][32], kb);
    const v16h vb0 = load_frag(&VsT[w * 16 + ml][0],  kb);
    const v16h vb1 = load_frag(&VsT[w * 16 + ml][32], kb);
    o = WMMA_F16(pa0, vb0, o);
    o = WMMA_F16(pa1, vb1, o);
    __syncthreads();
  }

  // Epilogue: O / Lrow -> global (wave w owns dims w*16..w*16+15)
  #pragma unroll
  for (int r = 0; r < 8; ++r) {
    const int m = kb + r;
    const float val = o[r] / Lrow[m];
    const size_t idx = base + (size_t)(q0r + m) * INNER_ + (size_t)(w * 16 + ml);
    if (accum) O[idx] += val;
    else       O[idx] = val;
  }
}

// ---------------------------------------------------------------------------
// Host-side orchestration
// ---------------------------------------------------------------------------
extern "C" void kernel_launch(void* const* d_in, const int* in_sizes, int n_in,
                              void* d_out, int out_size, void* d_ws, size_t ws_size,
                              hipStream_t stream)
{
  (void)in_sizes; (void)n_in; (void)out_size; (void)ws_size;

  const float* h_t    = (const float*)d_in[0];
  const float* h_a    = (const float*)d_in[1];
  const float* h_v    = (const float*)d_in[2];
  const float* W_q    = (const float*)d_in[3];
  const float* W_k_ta = (const float*)d_in[4];
  const float* W_k_tv = (const float*)d_in[5];
  const float* W_v_ta = (const float*)d_in[6];
  const float* W_v_tv = (const float*)d_in[7];
  const float* W_out  = (const float*)d_in[8];
  const float* b_out  = (const float*)d_in[9];
  float* out = (float*)d_out;

  // Workspace layout (bytes)
  char* ws = (char*)d_ws;
  const size_t MB = 1024 * 1024;
  _Float16* qh   = (_Float16*)(ws + 0   * MB);  // 5 x f16[4096,1024] = 40 MB
  _Float16* kta  = (_Float16*)(ws + 8   * MB);
  _Float16* vta  = (_Float16*)(ws + 16  * MB);
  _Float16* ktv  = (_Float16*)(ws + 24  * MB);
  _Float16* vtv  = (_Float16*)(ws + 32  * MB);
  float*    attn = (float*)   (ws + 40  * MB);  // f32[4096,1024] = 16 MB
  _Float16* at16 = (_Float16*)(ws + 56  * MB);  // f16[4096,1024] =  8 MB
  _Float16* ht16 = (_Float16*)(ws + 64  * MB);  // 3 x f16[4096,2048] = 48 MB
  _Float16* ha16 = (_Float16*)(ws + 80  * MB);
  _Float16* hv16 = (_Float16*)(ws + 96  * MB);
  _Float16* wqT  = (_Float16*)(ws + 112 * MB);  // 5 x f16[1024,2048] = 20 MB
  _Float16* wktaT= (_Float16*)(ws + 116 * MB);
  _Float16* wvtaT= (_Float16*)(ws + 120 * MB);
  _Float16* wktvT= (_Float16*)(ws + 124 * MB);
  _Float16* wvtvT= (_Float16*)(ws + 128 * MB);
  _Float16* woT  = (_Float16*)(ws + 132 * MB);  // f16[2048,1024] = 4 MB

  // --- Prep: fp32 -> f16 activations; transpose-convert weights to [N][K] ---
  const int nh4 = MTOT * DIMM / 4;
  ta_cvt_f32_f16<<<dim3((nh4 + 255) / 256), dim3(256), 0, stream>>>(h_t, ht16, nh4);
  ta_cvt_f32_f16<<<dim3((nh4 + 255) / 256), dim3(256), 0, stream>>>(h_a, ha16, nh4);
  ta_cvt_f32_f16<<<dim3((nh4 + 255) / 256), dim3(256), 0, stream>>>(h_v, hv16, nh4);

  const dim3 gtw(INNER_ / 64, DIMM / 64);       // (16, 32) for [2048][1024] weights
  ta_transpose_cvt<<<gtw, dim3(256), 0, stream>>>(W_q,    wqT,   DIMM, INNER_);
  ta_transpose_cvt<<<gtw, dim3(256), 0, stream>>>(W_k_ta, wktaT, DIMM, INNER_);
  ta_transpose_cvt<<<gtw, dim3(256), 0, stream>>>(W_v_ta, wvtaT, DIMM, INNER_);
  ta_transpose_cvt<<<gtw, dim3(256), 0, stream>>>(W_k_tv, wktvT, DIMM, INNER_);
  ta_transpose_cvt<<<gtw, dim3(256), 0, stream>>>(W_v_tv, wvtvT, DIMM, INNER_);
  ta_transpose_cvt<<<dim3(DIMM / 64, INNER_ / 64), dim3(256), 0, stream>>>(W_out, woT, INNER_, DIMM);

  // --- Projections: all-f16 async-staged GEMMs, f16 out ---
  const dim3 blk(512);
  const dim3 gproj(INNER_ / 128, MTOT / 128);   // (8, 32)
  ta_gemm_wmma_f16<<<gproj, blk, 0, stream>>>(ht16, wqT,   qh,  nullptr, nullptr, nullptr, MTOT, INNER_, DIMM);
  ta_gemm_wmma_f16<<<gproj, blk, 0, stream>>>(ha16, wktaT, kta, nullptr, nullptr, nullptr, MTOT, INNER_, DIMM);
  ta_gemm_wmma_f16<<<gproj, blk, 0, stream>>>(ha16, wvtaT, vta, nullptr, nullptr, nullptr, MTOT, INNER_, DIMM);
  ta_gemm_wmma_f16<<<gproj, blk, 0, stream>>>(hv16, wktvT, ktv, nullptr, nullptr, nullptr, MTOT, INNER_, DIMM);
  ta_gemm_wmma_f16<<<gproj, blk, 0, stream>>>(hv16, wvtvT, vtv, nullptr, nullptr, nullptr, MTOT, INNER_, DIMM);

  // --- Two cross-attentions; second accumulates into the first's output ---
  const dim3 gattn(NN / 16, HEADS_, BB);        // (128, 16, 2)
  ta_flash_attn_wmma<<<gattn, dim3(128), 0, stream>>>(qh, kta, vta, attn, 0, SCALE_F);
  ta_flash_attn_wmma<<<gattn, dim3(128), 0, stream>>>(qh, ktv, vtv, attn, 1, SCALE_F);

  // --- Output projection + residual + bias: fp32 out ---
  const int na4 = MTOT * INNER_ / 4;
  ta_cvt_f32_f16<<<dim3((na4 + 255) / 256), dim3(256), 0, stream>>>(attn, at16, na4);
  const dim3 gout(DIMM / 128, MTOT / 128);      // (16, 32)
  ta_gemm_wmma_f16<<<gout, blk, 0, stream>>>(at16, woT, nullptr, out, h_t, b_out, MTOT, DIMM, INNER_);
}
